// HyperbolicTemporalEncoder_53618371723382
// MI455X (gfx1250) — matrix operand order repaired and checked by reading, not compiled, not run on hardware
//
#include <hip/hip_runtime.h>
#include <hip/hip_bf16.h>
#include <math.h>

// MI455X / gfx1250: wave32, WMMA 16x16x32 bf16 -> f32 accumulate.
typedef __attribute__((ext_vector_type(16))) __bf16 v16bf;
typedef __attribute__((ext_vector_type(8)))  float  v8f;

__device__ __forceinline__ void fatomic_add(float* p, float v) {
  // lowers to global_atomic_add_f32 (no return), agent scope
  __hip_atomic_fetch_add(p, v, __ATOMIC_RELAXED, __HIP_MEMORY_SCOPE_AGENT);
}
__device__ __forceinline__ void fatomic_max(float* p, float v) {
  // sign-split trick: works for mixed-sign floats with -inf init
  if (v >= 0.0f) atomicMax((int*)p, __float_as_int(v));
  else           atomicMin((unsigned int*)p, __float_as_uint(v));
}

// ---------------- utility kernels ----------------
__global__ void k_fill(float* p, float v, int n) {
  int i = blockIdx.x * blockDim.x + threadIdx.x;
  if (i < n) p[i] = v;
}

__global__ void k_cvt_bf16(const float* in, __bf16* out, int n) {
  int i = blockIdx.x * blockDim.x + threadIdx.x;
  if (i < n) out[i] = (__bf16)in[i];
}

// ---------------- temporal gate ----------------
__global__ void k_edge_time(const int* __restrict__ src, const int* __restrict__ dst,
                            const float* __restrict__ t, float* nt, float* cnt, int E) {
  int e = blockIdx.x * blockDim.x + threadIdx.x;
  if (e >= E) return;
  int s = src[e], d = dst[e];
  float tv = t[e];
  fatomic_add(nt + s, tv);
  fatomic_add(nt + d, tv);
  fatomic_add(cnt + s, 1.0f);
  fatomic_add(cnt + d, 1.0f);
}

// node_time (in-place over nt) + global max/sum reduction into scal[0..1]
__global__ void k_node_time(float* nt, const float* cnt, float* scal, int N) {
  int i = blockIdx.x * blockDim.x + threadIdx.x;
  if (i >= N) return;
  float v = nt[i] / fmaxf(cnt[i], 1.0f);
  nt[i] = v;                       // node_time, always >= 0
  atomicMax((int*)&scal[0], __float_as_int(v));   // nonneg -> int-bit max ok
  fatomic_add(&scal[1], v);
}

// x + gate*W_time + b_time, emitted directly as bf16 GEMM operand
__global__ void k_apply_gate(const float* __restrict__ x, const float* __restrict__ node_time,
                             const float* __restrict__ scal, const float* __restrict__ braw,
                             const float* __restrict__ Wt, const float* __restrict__ bt,
                             __bf16* abuf, int N) {
  int idx = blockIdx.x * blockDim.x + threadIdx.x;
  if (idx >= N * 128) return;
  int i = idx >> 7, d = idx & 127;
  float maxv = scal[0];
  float meanv = scal[1] / (float)N;
  float scale = fmaxf(maxv - meanv, 1e-6f);
  float beta = log1pf(expf(braw[0])) + 1e-6f;     // softplus
  float delta = maxv - node_time[i];
  float gate = expf(-(beta * delta / scale));
  float v = x[idx] + gate * Wt[d] + bt[d];
  abuf[idx] = (__bf16)v;
}

// ---------------- WMMA GEMM: C[nrows x 128] = A[nrows x 128] @ W[128 x 128]^T ----------------
// Block = 8 waves = 128 rows. W (32KB bf16) staged once per block in LDS -> 8x reuse,
// B served by ds_load_b128. B-tiles preloaded per n-tile so the 4 WMMAs drain in a burst.
__global__ __launch_bounds__(256) void k_gemm_bf16(const __bf16* __restrict__ A,
                                                   const __bf16* __restrict__ W,
                                                   float* __restrict__ C,
                                                   int nrows) {
  __shared__ __bf16 wlds[128 * 128];           // 32 KB of the 320 KB WGP LDS
  int tid = threadIdx.x;
  // cooperative stage: 2048 x b128 copies, 8 per thread
  {
    const float4* wsrc = reinterpret_cast<const float4*>(W);
    float4* wdst = reinterpret_cast<float4*>(wlds);
#pragma unroll
    for (int i = 0; i < 8; ++i) wdst[tid + i * 256] = wsrc[tid + i * 256];
  }
  __syncthreads();

  int wv = tid >> 5, lane = tid & 31;
  int nrt = (nrows + 15) >> 4;
  int rt = blockIdx.x * 8 + wv;
  if (rt >= nrt) return;
  int half = lane >> 4, l15 = lane & 15;

  // A-tile, contiguous k order matching B: lane row = l15; e0..15 -> K = kt*32+16*half+(0..15) (contiguous)
  int rowA = rt * 16 + l15;
  if (rowA >= nrows) rowA = nrows - 1;
  v16bf a[4];
  const __bf16* ap = A + (size_t)rowA * 128 + half * 16;
#pragma unroll
  for (int kt = 0; kt < 4; ++kt) {
    float4* av = reinterpret_cast<float4*>(&a[kt]);
    av[0] = *reinterpret_cast<const float4*>(ap + kt * 32);
    av[1] = *reinterpret_cast<const float4*>(ap + kt * 32 + 8);
  }

  bool fullTile = (rt * 16 + 16) <= nrows;
#pragma unroll
  for (int tn = 0; tn < 8; ++tn) {
    // preload all 4 B-tiles from LDS: B[k][n] = W[n][k]; K run = kt*32 + 16*half + e (contiguous)
    const __bf16* wp = wlds + (size_t)(tn * 16 + l15) * 128 + half * 16;
    v16bf b[4];
#pragma unroll
    for (int kt = 0; kt < 4; ++kt) {
      float4* bv = reinterpret_cast<float4*>(&b[kt]);
      bv[0] = *reinterpret_cast<const float4*>(wp + kt * 32);
      bv[1] = *reinterpret_cast<const float4*>(wp + kt * 32 + 8);
    }
    v8f acc = {0.f, 0.f, 0.f, 0.f, 0.f, 0.f, 0.f, 0.f};
#pragma unroll
    for (int kt = 0; kt < 4; ++kt)
      acc = __builtin_amdgcn_wmma_f32_16x16x32_bf16(false, a[kt], false, b[kt],
                                                    (short)0, acc, false, false);
    // C/D layout: lanes16-31 -> M+8; store rows unpredicated in the common full-tile case
    float* crow = C + (size_t)(rt * 16 + half * 8) * 128 + tn * 16 + l15;
    if (fullTile) {
#pragma unroll
      for (int r = 0; r < 8; ++r) crow[r * 128] = acc[r];
    } else {
#pragma unroll
      for (int r = 0; r < 8; ++r)
        if (rt * 16 + half * 8 + r < nrows) crow[r * 128] = acc[r];
    }
  }
}

// ---------------- attention helpers ----------------
// wave per node: hs = h.a_s, hd = h.a_d
__global__ void k_node_dots(const float* __restrict__ h, const float* __restrict__ as_,
                            const float* __restrict__ ad_, float* hs, float* hd, int N) {
  int gw = (blockIdx.x * blockDim.x + threadIdx.x) >> 5;
  int lane = threadIdx.x & 31;
  if (gw >= N) return;
  float4 hv = reinterpret_cast<const float4*>(h + (size_t)gw * 128)[lane];
  float4 a1 = reinterpret_cast<const float4*>(as_)[lane];
  float4 a2 = reinterpret_cast<const float4*>(ad_)[lane];
  float ps = hv.x * a1.x + hv.y * a1.y + hv.z * a1.z + hv.w * a1.w;
  float pd = hv.x * a2.x + hv.y * a2.y + hv.z * a2.z + hv.w * a2.w;
  for (int off = 16; off; off >>= 1) {
    ps += __shfl_xor(ps, off, 32);
    pd += __shfl_xor(pd, off, 32);
  }
  if (lane == 0) { hs[gw] = ps; hd[gw] = pd; }
}

__device__ __forceinline__ void edge_sd(int e, const int* src, const int* dst, int E,
                                        int& s, int& d) {
  if (e < E) { s = src[e]; d = dst[e]; } else { s = e - E; d = s; }  // self-loops appended
}

__global__ void k_edge_max(const int* __restrict__ src, const int* __restrict__ dst,
                           const float* __restrict__ hs, const float* __restrict__ hd,
                           float* emax, int E, int N) {
  int e = blockIdx.x * blockDim.x + threadIdx.x;
  if (e >= E + N) return;
  int s, d; edge_sd(e, src, dst, E, s, d);
  float sc = hs[s] + hd[d];
  sc = sc > 0.f ? sc : 0.2f * sc;               // leaky_relu(0.2)
  fatomic_max(emax + d, sc);
}

__global__ void k_edge_expsum(const int* __restrict__ src, const int* __restrict__ dst,
                              const float* __restrict__ hs, const float* __restrict__ hd,
                              const float* __restrict__ emax, float* pbuf, float* denom,
                              int E, int N) {
  int e = blockIdx.x * blockDim.x + threadIdx.x;
  if (e >= E + N) return;
  int s, d; edge_sd(e, src, dst, E, s, d);
  float sc = hs[s] + hd[d];
  sc = sc > 0.f ? sc : 0.2f * sc;
  float m = emax[d];
  if (!__builtin_isfinite(m)) m = 0.f;
  float p = expf(sc - m);
  pbuf[e] = p;
  fatomic_add(denom + d, p);
}

// wave per edge: out[dst] += alpha * h[src], 4 floats per lane
__global__ void k_edge_agg(const int* __restrict__ src, const int* __restrict__ dst,
                           const float* __restrict__ pbuf, const float* __restrict__ denom,
                           const float* __restrict__ h, float* out, int E, int N) {
  int gw = (blockIdx.x * blockDim.x + threadIdx.x) >> 5;
  int lane = threadIdx.x & 31;
  if (gw >= E + N) return;
  int s, d; edge_sd(gw, src, dst, E, s, d);
  float alpha = pbuf[gw] / fmaxf(denom[d], 1e-16f);
  float4 hv = reinterpret_cast<const float4*>(h + (size_t)s * 128)[lane];
  float* op = out + (size_t)d * 128 + lane * 4;
  fatomic_add(op + 0, alpha * hv.x);
  fatomic_add(op + 1, alpha * hv.y);
  fatomic_add(op + 2, alpha * hv.z);
  fatomic_add(op + 3, alpha * hv.w);
}

template <bool RELU>
__global__ void k_bias_act(float* buf, const float* __restrict__ b, __bf16* abuf, int total) {
  int idx = blockIdx.x * blockDim.x + threadIdx.x;
  if (idx >= total) return;
  int d = idx & 127;
  float v = buf[idx] + b[d];
  if (RELU) v = fmaxf(v, 0.f);
  buf[idx] = v;
  abuf[idx] = (__bf16)v;                         // stage next GEMM operand
}

// gh = W_hh @ init_hidden + b_hh  (single 384-vector, hist is a broadcast)
__global__ void k_gh(const float* __restrict__ Whh, const float* __restrict__ bhh,
                     const float* __restrict__ h0, float* gh) {
  int j = blockIdx.x * blockDim.x + threadIdx.x;
  if (j >= 384) return;
  float a = bhh[j];
  const float* wr = Whh + (size_t)j * 128;
  for (int k = 0; k < 128; ++k) a += wr[k] * h0[k];
  gh[j] = a;
}

// ---------------- fused GRU GEMM + expmap0 + ball projection ----------------
// block = 16 rows; 8 waves cover 24 n-tiles (384 cols) via WMMA; gi staged in LDS (24KB).
// Wave n-tiles are disjoint (no W reuse within block) -> W stays in global/L0; B-tiles
// preloaded per n-tile so the 4 WMMAs issue as a burst.
__global__ __launch_bounds__(256) void k_gru_expmap(const __bf16* __restrict__ A,
                                                    const __bf16* __restrict__ Wih,
                                                    const float* __restrict__ bih,
                                                    const float* __restrict__ gh,
                                                    const float* __restrict__ h0,
                                                    float* __restrict__ out, int nrows) {
  __shared__ float gil[16 * 384];
  int wv = threadIdx.x >> 5, lane = threadIdx.x & 31;
  int rt = blockIdx.x;
  int half = lane >> 4, l15 = lane & 15;

  int rowA = rt * 16 + l15;
  if (rowA >= nrows) rowA = nrows - 1;
  v16bf a[4];
  const __bf16* ap = A + (size_t)rowA * 128 + half * 16;
#pragma unroll
  for (int kt = 0; kt < 4; ++kt) {
    float4* av = reinterpret_cast<float4*>(&a[kt]);
    av[0] = *reinterpret_cast<const float4*>(ap + kt * 32);
    av[1] = *reinterpret_cast<const float4*>(ap + kt * 32 + 8);
  }

#pragma unroll
  for (int ti = 0; ti < 3; ++ti) {
    int tn = wv * 3 + ti;
    const __bf16* wp = Wih + (size_t)(tn * 16 + l15) * 128 + half * 16;
    v16bf b[4];
#pragma unroll
    for (int kt = 0; kt < 4; ++kt) {
      float4* bv = reinterpret_cast<float4*>(&b[kt]);
      bv[0] = *reinterpret_cast<const float4*>(wp + kt * 32);
      bv[1] = *reinterpret_cast<const float4*>(wp + kt * 32 + 8);
    }
    v8f acc = {0.f, 0.f, 0.f, 0.f, 0.f, 0.f, 0.f, 0.f};
#pragma unroll
    for (int kt = 0; kt < 4; ++kt)
      acc = __builtin_amdgcn_wmma_f32_16x16x32_bf16(false, a[kt], false, b[kt],
                                                    (short)0, acc, false, false);
#pragma unroll
    for (int r = 0; r < 8; ++r)
      gil[(half * 8 + r) * 384 + tn * 16 + l15] = acc[r];
  }
  __syncthreads();

  // epilogue: 16 threads per row, 8 dims each
  int row = threadIdx.x >> 4;
  int j = threadIdx.x & 15;
  int grow = rt * 16 + row;
  float ht[8];
  float ss = 0.f;
#pragma unroll
  for (int q = 0; q < 8; ++q) {
    int d = j * 8 + q;
    float gir = gil[row * 384 + d]       + bih[d]       + gh[d];
    float giz = gil[row * 384 + 128 + d] + bih[128 + d] + gh[128 + d];
    float gin = gil[row * 384 + 256 + d] + bih[256 + d];
    float r_ = 1.f / (1.f + expf(-gir));
    float z_ = 1.f / (1.f + expf(-giz));
    float n_ = tanhf(gin + r_ * gh[256 + d]);
    float v = (1.f - z_) * n_ + z_ * h0[d];
    ht[q] = v;
    ss += v * v;
  }
  for (int off = 8; off; off >>= 1) ss += __shfl_xor(ss, off, 16);
  float norm = fmaxf(sqrtf(ss), 1e-12f);         // sqrt(c)=1
  float th = tanhf(norm);
  float f = th / norm;                           // expmap0 scale
  f *= fminf((1.0f - 1e-5f) / fmaxf(th, 1e-12f), 1.0f);  // project_to_ball
  if (grow < nrows) {
#pragma unroll
    for (int q = 0; q < 8; ++q) out[(size_t)grow * 128 + j * 8 + q] = ht[q] * f;
  }
}

// ---------------- launcher ----------------
extern "C" void kernel_launch(void* const* d_in, const int* in_sizes, int n_in,
                              void* d_out, int out_size, void* d_ws, size_t ws_size,
                              hipStream_t stream) {
  const float* x      = (const float*)d_in[0];
  const float* t      = (const float*)d_in[1];
  const float* W_time = (const float*)d_in[2];
  const float* b_time = (const float*)d_in[3];
  const float* W1     = (const float*)d_in[4];
  const float* a_src1 = (const float*)d_in[5];
  const float* a_dst1 = (const float*)d_in[6];
  const float* b1     = (const float*)d_in[7];
  const float* W2     = (const float*)d_in[8];
  const float* a_src2 = (const float*)d_in[9];
  const float* a_dst2 = (const float*)d_in[10];
  const float* b2     = (const float*)d_in[11];
  const float* W_ih   = (const float*)d_in[12];
  const float* W_hh   = (const float*)d_in[13];
  const float* b_ih   = (const float*)d_in[14];
  const float* b_hh   = (const float*)d_in[15];
  const float* h0     = (const float*)d_in[16];
  const float* braw   = (const float*)d_in[17];
  const int*   eidx   = (const int*)d_in[18];

  const int N = in_sizes[0] / 128;
  const int E = in_sizes[1];
  const int* src = eidx;
  const int* dst = eidx + E;

  char* w = (char*)d_ws;
  size_t off = 0;
  auto alloc = [&](size_t bytes) -> char* {
    off = (off + 255) & ~(size_t)255;
    char* p = w + off;
    off += bytes;
    return p;
  };
  float*  nt    = (float*)alloc((size_t)N * 4);
  float*  cnt   = (float*)alloc((size_t)N * 4);
  float*  scal  = (float*)alloc(16);
  float*  hs    = (float*)alloc((size_t)N * 4);
  float*  hd    = (float*)alloc((size_t)N * 4);
  float*  emax  = (float*)alloc((size_t)N * 4);
  float*  denom = (float*)alloc((size_t)N * 4);
  float*  pbuf  = (float*)alloc((size_t)(E + N) * 4);
  float*  gh    = (float*)alloc(384 * 4);
  __bf16* abuf  = (__bf16*)alloc((size_t)N * 128 * 2);
  __bf16* wb1   = (__bf16*)alloc(128 * 128 * 2);
  __bf16* wb2   = (__bf16*)alloc(128 * 128 * 2);
  __bf16* wbih  = (__bf16*)alloc(384 * 128 * 2);
  float*  hA    = (float*)alloc((size_t)N * 128 * 4);
  float*  hB    = (float*)alloc((size_t)N * 128 * 4);

  auto cdiv = [](long a, long b) { return (int)((a + b - 1) / b); };
  const int T = 256;
  const long ND = (long)N * 128;
  const int nrt = cdiv(N, 16);
  const int NE = E + N;

  // weight bf16 staging
  k_cvt_bf16<<<cdiv(128 * 128, T), T, 0, stream>>>(W1, wb1, 128 * 128);
  k_cvt_bf16<<<cdiv(128 * 128, T), T, 0, stream>>>(W2, wb2, 128 * 128);
  k_cvt_bf16<<<cdiv(384 * 128, T), T, 0, stream>>>(W_ih, wbih, 384 * 128);

  // temporal gate
  k_fill<<<cdiv(N, T), T, 0, stream>>>(nt, 0.f, N);
  k_fill<<<cdiv(N, T), T, 0, stream>>>(cnt, 0.f, N);
  k_fill<<<1, 4, 0, stream>>>(scal, 0.f, 4);
  k_edge_time<<<cdiv(E, T), T, 0, stream>>>(src, dst, t, nt, cnt, E);
  k_node_time<<<cdiv(N, T), T, 0, stream>>>(nt, cnt, scal, N);
  k_apply_gate<<<cdiv(ND, T), T, 0, stream>>>(x, nt, scal, braw, W_time, b_time, abuf, N);

  // ---- GAT layer 1 ----
  k_gemm_bf16<<<cdiv(nrt, 8), T, 0, stream>>>(abuf, wb1, hA, N);
  k_node_dots<<<cdiv(N, 8), T, 0, stream>>>(hA, a_src1, a_dst1, hs, hd, N);
  k_fill<<<cdiv(N, T), T, 0, stream>>>(emax, -INFINITY, N);
  k_fill<<<cdiv(N, T), T, 0, stream>>>(denom, 0.f, N);
  k_fill<<<cdiv(ND, T), T, 0, stream>>>(hB, 0.f, (int)ND);
  k_edge_max<<<cdiv(NE, T), T, 0, stream>>>(src, dst, hs, hd, emax, E, N);
  k_edge_expsum<<<cdiv(NE, T), T, 0, stream>>>(src, dst, hs, hd, emax, pbuf, denom, E, N);
  k_edge_agg<<<cdiv(NE, 8), T, 0, stream>>>(src, dst, pbuf, denom, hA, hB, E, N);
  k_bias_act<true><<<cdiv(ND, T), T, 0, stream>>>(hB, b1, abuf, (int)ND);

  // ---- GAT layer 2 ----
  k_gemm_bf16<<<cdiv(nrt, 8), T, 0, stream>>>(abuf, wb2, hA, N);
  k_node_dots<<<cdiv(N, 8), T, 0, stream>>>(hA, a_src2, a_dst2, hs, hd, N);
  k_fill<<<cdiv(N, T), T, 0, stream>>>(emax, -INFINITY, N);
  k_fill<<<cdiv(N, T), T, 0, stream>>>(denom, 0.f, N);
  k_fill<<<cdiv(ND, T), T, 0, stream>>>(hB, 0.f, (int)ND);
  k_edge_max<<<cdiv(NE, T), T, 0, stream>>>(src, dst, hs, hd, emax, E, N);
  k_edge_expsum<<<cdiv(NE, T), T, 0, stream>>>(src, dst, hs, hd, emax, pbuf, denom, E, N);
  k_edge_agg<<<cdiv(NE, 8), T, 0, stream>>>(src, dst, pbuf, denom, hA, hB, E, N);
  k_bias_act<false><<<cdiv(ND, T), T, 0, stream>>>(hB, b2, abuf, (int)ND);

  // ---- fused GRU + expmap0 + projection ----
  k_gh<<<cdiv(384, T), T, 0, stream>>>(W_hh, b_hh, h0, gh);
  k_gru_expmap<<<nrt, T, 0, stream>>>(abuf, wbih, b_ih, gh, h0, (float*)d_out, N);
}
